// SmoothFakeDynamicallyQuantizedLinear_11158325035112
// MI455X (gfx1250) — 8-bit matrix operand order repaired and checked by reading, compiled, not run on hardware
//
#include <hip/hip_runtime.h>
#include <hip/hip_bf16.h>


typedef int v8i __attribute__((ext_vector_type(8)));

union V8U {
    v8i  v;
    int2 d2[4];
    int4 d4[2];
};

struct ABuf { V8U a[2]; };
struct BBuf { V8U b[4]; };

#define FLT_EPS_F 1.19209290e-07f

__device__ inline unsigned lv(unsigned x) {
    asm volatile("" : "+v"(x));
    return x;
}

__device__ inline float block_absmax(float v) {
    #pragma unroll
    for (int off = 16; off > 0; off >>= 1)
        v = fmaxf(v, __shfl_xor(v, off));
    __shared__ float red[8];
    const int lane = threadIdx.x & 31;
    const int wv   = threadIdx.x >> 5;
    if (lane == 0) red[wv] = v;
    __syncthreads();
    if (wv == 0) {
        float t = (lane < 8) ? red[lane] : 0.0f;
        #pragma unroll
        for (int off = 4; off > 0; off >>= 1)
            t = fmaxf(t, __shfl_xor(t, off));
        if (lane == 0) red[0] = t;
    }
    __syncthreads();
    float r = red[0];
    __syncthreads();
    return r;
}

__device__ inline int clip_round(float v, float lo, float hi) {
    v = __builtin_rintf(v);
    v = fminf(fmaxf(v, lo), hi);
    return (int)v;
}

__global__ __launch_bounds__(256) void quant_w_kernel(
    const float* __restrict__ W, const float* __restrict__ ss,
    char* __restrict__ Wq, float* __restrict__ wscale, int K)
{
    const int o   = blockIdx.x;
    const int tid = threadIdx.x;
    const float4* wr = (const float4*)(W + (size_t)o * K) + tid * 4;
    const float4* sr = (const float4*)(ss) + tid * 4;

    float vals[16];
    float amax = 0.0f;
    #pragma unroll
    for (int i = 0; i < 4; ++i) {
        float4 w = wr[i];
        float4 s = sr[i];
        vals[4*i+0] = w.x * s.x;
        vals[4*i+1] = w.y * s.y;
        vals[4*i+2] = w.z * s.z;
        vals[4*i+3] = w.w * s.w;
        amax = fmaxf(amax, fabsf(vals[4*i+0]));
        amax = fmaxf(amax, fabsf(vals[4*i+1]));
        amax = fmaxf(amax, fabsf(vals[4*i+2]));
        amax = fmaxf(amax, fabsf(vals[4*i+3]));
    }
    amax = block_absmax(amax);
    const float scale = fmaxf(amax / 127.5f, FLT_EPS_F);
    if (tid == 0) wscale[o] = scale;
    const float inv = 1.0f / scale;

    int4 packed;
    int* pk = (int*)&packed;
    #pragma unroll
    for (int g = 0; g < 4; ++g) {
        int b0 = clip_round(vals[4*g+0] * inv, -128.f, 127.f);
        int b1 = clip_round(vals[4*g+1] * inv, -128.f, 127.f);
        int b2 = clip_round(vals[4*g+2] * inv, -128.f, 127.f);
        int b3 = clip_round(vals[4*g+3] * inv, -128.f, 127.f);
        pk[g] = (b0 & 255) | ((b1 & 255) << 8) | ((b2 & 255) << 16) | ((b3 & 255) << 24);
    }
    ((int4*)(Wq + (size_t)o * K))[tid] = packed;
}

__global__ __launch_bounds__(256) void quant_x_kernel(
    const float* __restrict__ X, const float* __restrict__ ss,
    char* __restrict__ Xq, float* __restrict__ xscale, int K)
{
    const int t   = blockIdx.x;
    const int tid = threadIdx.x;
    const float4* xr = (const float4*)(X + (size_t)t * K) + tid * 4;
    const float4* sr = (const float4*)(ss) + tid * 4;

    float vals[16];
    float amax = 0.0f;
    #pragma unroll
    for (int i = 0; i < 4; ++i) {
        float4 x = xr[i];
        float4 s = sr[i];
        vals[4*i+0] = x.x / s.x;
        vals[4*i+1] = x.y / s.y;
        vals[4*i+2] = x.z / s.z;
        vals[4*i+3] = x.w / s.w;
        amax = fmaxf(amax, fabsf(vals[4*i+0]));
        amax = fmaxf(amax, fabsf(vals[4*i+1]));
        amax = fmaxf(amax, fabsf(vals[4*i+2]));
        amax = fmaxf(amax, fabsf(vals[4*i+3]));
    }
    amax = block_absmax(amax);
    const float scale = fmaxf(amax, 1e-5f) / 127.0f;
    if (tid == 0) xscale[t] = scale;
    const float inv = 1.0f / scale;

    int4 packed;
    int* pk = (int*)&packed;
    #pragma unroll
    for (int g = 0; g < 4; ++g) {
        int b0 = clip_round(vals[4*g+0] * inv, -127.f, 127.f);
        int b1 = clip_round(vals[4*g+1] * inv, -127.f, 127.f);
        int b2 = clip_round(vals[4*g+2] * inv, -127.f, 127.f);
        int b3 = clip_round(vals[4*g+3] * inv, -127.f, 127.f);
        pk[g] = (b0 & 255) | ((b1 & 255) << 8) | ((b2 & 255) << 16) | ((b3 & 255) << 24);
    }
    ((int4*)(Xq + (size_t)t * K))[tid] = packed;
}

__device__ inline void load_a_tile(V8U& a, const char* base  , unsigned voff) {
    const char* p = base + (size_t)voff;
    a.d4[0] = *(const int4*)(p +  0);
    a.d4[1] = *(const int4*)(p + 32);
}

__device__ inline void load_b_tile(V8U& b, const char* base  , unsigned voff) {
    const char* p = base + (size_t)voff;
    b.d4[0] = *(const int4*)(p +  0);
    b.d4[1] = *(const int4*)(p + 32);
}

__device__ inline void dot8(v8i acc[8], const ABuf& A, const BBuf& B) {
    #pragma unroll
    for (int mt = 0; mt < 2; ++mt)
        #pragma unroll
        for (int nt = 0; nt < 4; ++nt)
            acc[mt * 4 + nt] = __builtin_amdgcn_wmma_i32_16x16x64_iu8(
                true, A.a[mt].v, true, B.b[nt].v, acc[mt * 4 + nt],
                 false,  false);
}

__global__ __launch_bounds__(256) void gemm_i8_wmma_kernel(
    const char* __restrict__ Xq, const char* __restrict__ Wq,
    const float* __restrict__ xscale, const float* __restrict__ wscale,
    const float* __restrict__ bias, float* __restrict__ Y,
    int M, int N, int K)
{
    const int lane = threadIdx.x & 31;
    const int wave = threadIdx.x >> 5;
    const int half = lane >> 4;
    const int l16  = lane & 15;
    const int wm   = wave >> 2;
    const int wn   = wave & 3;

    const int m0 = blockIdx.y * 64  + wm * 32;
    const int n0 = blockIdx.x * 256 + wn * 64;

    const unsigned uK     = (unsigned)K;
    const unsigned aOff0c = (unsigned)(m0 +      l16) * uK + (unsigned)(half * 16);
    const unsigned aOff1c = (unsigned)(m0 + 16 + l16) * uK + (unsigned)(half * 16);
    const unsigned bOff0c = (unsigned)(n0 + 0*16 + l16) * uK + (unsigned)(half * 16);
    const unsigned bOff1c = (unsigned)(n0 + 1*16 + l16) * uK + (unsigned)(half * 16);
    const unsigned bOff2c = (unsigned)(n0 + 2*16 + l16) * uK + (unsigned)(half * 16);
    const unsigned bOff3c = (unsigned)(n0 + 3*16 + l16) * uK + (unsigned)(half * 16);

    v8i acc[8] = {};

    ABuf A0, A1;
    BBuf B0, B1;
    load_a_tile(A0.a[0], Xq, aOff0c);
    load_a_tile(A0.a[1], Xq, aOff1c);
    load_b_tile(B0.b[0], Wq, bOff0c);
    load_b_tile(B0.b[1], Wq, bOff1c);
    load_b_tile(B0.b[2], Wq, bOff2c);
    load_b_tile(B0.b[3], Wq, bOff3c);

    for (int kk = 0; kk < K; kk += 128) {
        const int k1 = kk + 64;
        const int k2 = (kk + 128 < K) ? (kk + 128) : 0;
        const char* xk1 = Xq + k1;
        const char* wk1 = Wq + k1;
        const char* xk2 = Xq + k2;
        const char* wk2 = Wq + k2;

        const unsigned a0 = lv(aOff0c);
        const unsigned a1 = lv(aOff1c);
        const unsigned b0 = lv(bOff0c);
        const unsigned b1 = lv(bOff1c);
        const unsigned b2 = lv(bOff2c);
        const unsigned b3 = lv(bOff3c);

        __builtin_prefetch(xk1 + 960 + (size_t)a0, 0, 1);
        __builtin_prefetch(xk1 + 960 + (size_t)a1, 0, 1);

        load_a_tile(A1.a[0], xk1, a0);
        load_a_tile(A1.a[1], xk1, a1);
        load_b_tile(B1.b[0], wk1, b0);
        load_b_tile(B1.b[1], wk1, b1);
        load_b_tile(B1.b[2], wk1, b2);
        load_b_tile(B1.b[3], wk1, b3);
        dot8(acc, A0, B0);

        load_a_tile(A0.a[0], xk2, a0);
        load_a_tile(A0.a[1], xk2, a1);
        load_b_tile(B0.b[0], wk2, b0);
        load_b_tile(B0.b[1], wk2, b1);
        load_b_tile(B0.b[2], wk2, b2);
        load_b_tile(B0.b[3], wk2, b3);
        dot8(acc, A1, B1);
    }

    #pragma unroll
    for (int mt = 0; mt < 2; ++mt) {
        float xsv[8];
        #pragma unroll
        for (int r = 0; r < 8; ++r)
            xsv[r] = xscale[m0 + mt * 16 + r + half * 8];
        #pragma unroll
        for (int nt = 0; nt < 4; ++nt) {
            const int col  = n0 + nt * 16 + l16;
            const float wsv = wscale[col];
            const float bv  = bias[col];
            #pragma unroll
            for (int r = 0; r < 8; ++r) {
                const int row = m0 + mt * 16 + r + half * 8;
                Y[(size_t)row * N + col] =
                    (float)acc[mt * 4 + nt][r] * xsv[r] * wsv + bv;
            }
        }
    }
}

extern "C" void kernel_launch(void* const* d_in, const int* in_sizes, int n_in,
                              void* d_out, int out_size, void* d_ws, size_t ws_size,
                              hipStream_t stream) {
    const float* X    = (const float*)d_in[0];
    const float* W    = (const float*)d_in[1];
    const float* bias = (const float*)d_in[2];
    const float* ss   = (const float*)d_in[3];

    const int DIN  = in_sizes[3];
    const int DOUT = in_sizes[2];
    const int M    = in_sizes[0] / DIN;

    char*  Xq     = (char*)d_ws;
    char*  Wq     = Xq + (size_t)M * DIN;
    float* xscale = (float*)(Wq + (size_t)DOUT * DIN);
    float* wscale = xscale + M;
    float* Y      = (float*)d_out;

    quant_w_kernel<<<DOUT, 256, 0, stream>>>(W, ss, Wq, wscale, DIN);
    quant_x_kernel<<<M,    256, 0, stream>>>(X, ss, Xq, xscale, DIN);

    dim3 grid(DOUT / 256, M / 64);
    gemm_i8_wmma_kernel<<<grid, 256, 0, stream>>>(Xq, Wq, xscale, wscale, bias, Y,
                                                  M, DOUT, DIN);
}
